// LoRALayer_36567351558281
// MI455X (gfx1250) — matrix operand order repaired and checked by reading, not compiled, not run on hardware
//
#include <hip/hip_runtime.h>

// ---------------------------------------------------------------------------
// LoRA linear: out = x @ W^T + b + (x @ lora_b) @ lora_a * SCALE
//   x: (16384, 4096) f32   W: (4096, 4096) f32   b: (4096)
//   lora_a: (8, 4096)      lora_b: (4096, 8)
// Main GEMM: bf16x3 (truncating hi/lo split) on v_wmma_f32_16x16x32_bf16,
// LDS double-buffered, one barrier per K-step.
// ---------------------------------------------------------------------------

typedef __attribute__((ext_vector_type(16))) __bf16         v16bf;
typedef __attribute__((ext_vector_type(8)))  __bf16         v8bf;
typedef __attribute__((ext_vector_type(8)))  float          v8f;

#define KDIM   4096
#define NDIM   4096
#define RANK   8
#define LSCALE 2.0f            // alpha/rank = 16/8

#define BM   128
#define BN   128
#define BK   32
#define LDT  40                // padded LDS row stride (bf16 elems) -> 80B, conflict-free
#define NKT  (KDIM / BK)       // 128 K-steps
#define PLANE (BM * LDT)       // one buffer plane, in ushorts

__device__ __forceinline__ v8f wmma_bf16(v16bf a, v16bf b, v8f c) {
    // (neg_a, A, neg_b, B, c_mod, C, reuse_a, reuse_b)
    return __builtin_amdgcn_wmma_f32_16x16x32_bf16(false, a, false, b, (short)0, c,
                                                   false, false);
}

union Frag16 { v16bf v; v8bf h[2]; };

// Pack the high 16 bits of two floats into one dword: {f1[31:16], f0[31:16]}
__device__ __forceinline__ unsigned int pack_hi(unsigned int u0, unsigned int u1) {
    return __builtin_amdgcn_perm(u1, u0, 0x07060302u);   // one v_perm_b32
}

// ---------------------------------------------------------------------------
// Kernel 1: t[m, r] = sum_k x[m, k] * lora_b[k, r]     (one wave per row)
// ---------------------------------------------------------------------------
__global__ __launch_bounds__(256) void lora_t_kernel(const float* __restrict__ X,
                                                     const float* __restrict__ LB,
                                                     float* __restrict__ T) {
    const int lane = threadIdx.x & 31;
    const int wave = threadIdx.x >> 5;
    const int m    = blockIdx.x * 8 + wave;

    const float* xr = X + (size_t)m * KDIM;
    float acc[RANK];
#pragma unroll
    for (int r = 0; r < RANK; ++r) acc[r] = 0.0f;

#pragma unroll 4
    for (int it = 0; it < KDIM / 128; ++it) {
        const int k = it * 128 + lane * 4;
        const float4 xv = *(const float4*)(xr + k);
        const float xs[4] = {xv.x, xv.y, xv.z, xv.w};
#pragma unroll
        for (int kk = 0; kk < 4; ++kk) {
            const float4 b0 = *(const float4*)(LB + (size_t)(k + kk) * RANK);
            const float4 b1 = *(const float4*)(LB + (size_t)(k + kk) * RANK + 4);
            acc[0] += xs[kk] * b0.x;  acc[1] += xs[kk] * b0.y;
            acc[2] += xs[kk] * b0.z;  acc[3] += xs[kk] * b0.w;
            acc[4] += xs[kk] * b1.x;  acc[5] += xs[kk] * b1.y;
            acc[6] += xs[kk] * b1.z;  acc[7] += xs[kk] * b1.w;
        }
    }
#pragma unroll
    for (int off = 16; off > 0; off >>= 1)
#pragma unroll
        for (int r = 0; r < RANK; ++r) acc[r] += __shfl_xor(acc[r], off, 32);

    if (lane == 0) {
        float4* tp = (float4*)(T + (size_t)m * RANK);
        tp[0] = make_float4(acc[0], acc[1], acc[2], acc[3]);
        tp[1] = make_float4(acc[4], acc[5], acc[6], acc[7]);
    }
}

// ---------------------------------------------------------------------------
// Kernel 2: C = X * W^T (bf16x3 WMMA) + bias + SCALE * (T @ lora_a)
// block: 256 threads = 8 waves; block tile 128x128; wave tile 64x32 (4x2 WMMA)
// ---------------------------------------------------------------------------
__global__ __launch_bounds__(256) void lora_gemm_kernel(
    const float* __restrict__ X, const float* __restrict__ W,
    const float* __restrict__ bias, const float* __restrict__ lora_a,
    const float* __restrict__ T, float* __restrict__ out) {

    __shared__ alignas(16) unsigned short sAhi[2 * PLANE];
    __shared__ alignas(16) unsigned short sAlo[2 * PLANE];
    __shared__ alignas(16) unsigned short sBhi[2 * PLANE];
    __shared__ alignas(16) unsigned short sBlo[2 * PLANE];

    const int tid   = threadIdx.x;
    const int lane  = tid & 31;
    const int wave  = tid >> 5;
    const int waveN = wave & 3;          // 4 waves across N (32 cols each)
    const int waveM = wave >> 2;         // 2 waves across M (64 rows each)
    const int n0    = blockIdx.x * BN;
    const int m0    = blockIdx.y * BM;

    // global->LDS staging: thread (row = tid/2, 16-float chunk = tid&1)
    const int gRow = tid >> 1;
    const int gK   = (tid & 1) * 16;
    const float* aPtr = X + (size_t)(m0 + gRow) * KDIM + gK;
    const float* bPtr = W + (size_t)(n0 + gRow) * KDIM + gK;
    const int sOff = gRow * LDT + gK;

    v8f acc[4][2];
#pragma unroll
    for (int i = 0; i < 4; ++i)
#pragma unroll
        for (int j = 0; j < 2; ++j)
#pragma unroll
            for (int v = 0; v < 8; ++v) acc[i][j][v] = 0.0f;

    float4 ra[4], rb[4];

    auto load_regs = [&](int kt) {
        const float* ap = aPtr + (size_t)kt * BK;
        const float* bp = bPtr + (size_t)kt * BK;
#pragma unroll
        for (int q = 0; q < 4; ++q) {
            ra[q] = *(const float4*)(ap + 4 * q);
            rb[q] = *(const float4*)(bp + 4 * q);
        }
    };

    // Truncating hi/lo split + v_perm packing: 3 VALU per element.
    auto stash = [&](int buf) {
        unsigned int hA[8], lA[8], hB[8], lB[8];
#pragma unroll
        for (int p = 0; p < 8; ++p) {
            const float fa0 = ((const float*)ra)[2 * p];
            const float fa1 = ((const float*)ra)[2 * p + 1];
            const unsigned int ua0 = __float_as_uint(fa0);
            const unsigned int ua1 = __float_as_uint(fa1);
            hA[p] = pack_hi(ua0, ua1);
            const float la0 = fa0 - __uint_as_float(ua0 & 0xFFFF0000u);
            const float la1 = fa1 - __uint_as_float(ua1 & 0xFFFF0000u);
            lA[p] = pack_hi(__float_as_uint(la0), __float_as_uint(la1));

            const float fb0 = ((const float*)rb)[2 * p];
            const float fb1 = ((const float*)rb)[2 * p + 1];
            const unsigned int ub0 = __float_as_uint(fb0);
            const unsigned int ub1 = __float_as_uint(fb1);
            hB[p] = pack_hi(ub0, ub1);
            const float lb0 = fb0 - __uint_as_float(ub0 & 0xFFFF0000u);
            const float lb1 = fb1 - __uint_as_float(ub1 & 0xFFFF0000u);
            lB[p] = pack_hi(__float_as_uint(lb0), __float_as_uint(lb1));
        }
        const int off = buf * PLANE + sOff;
        *(uint4*)&sAhi[off]     = make_uint4(hA[0], hA[1], hA[2], hA[3]);
        *(uint4*)&sAhi[off + 8] = make_uint4(hA[4], hA[5], hA[6], hA[7]);
        *(uint4*)&sAlo[off]     = make_uint4(lA[0], lA[1], lA[2], lA[3]);
        *(uint4*)&sAlo[off + 8] = make_uint4(lA[4], lA[5], lA[6], lA[7]);
        *(uint4*)&sBhi[off]     = make_uint4(hB[0], hB[1], hB[2], hB[3]);
        *(uint4*)&sBhi[off + 8] = make_uint4(hB[4], hB[5], hB[6], hB[7]);
        *(uint4*)&sBlo[off]     = make_uint4(lB[0], lB[1], lB[2], lB[3]);
        *(uint4*)&sBlo[off + 8] = make_uint4(lB[4], lB[5], lB[6], lB[7]);
    };

    const int half = lane >> 4;          // 0: K-low half, 1: K-high half
    const int ln   = lane & 15;
    const int aRowBase = waveM * 64 + ln;
    const int bRowBase = waveN * 32 + ln;

    // prologue: stage K-step 0 into buffer 0
    load_regs(0);
    stash(0);
    __syncthreads();

    for (int kt = 0; kt < NKT; ++kt) {
        const int cur = kt & 1;
        const bool more = (kt + 1 < NKT);
        if (more) load_regs(kt + 1);     // global loads overlap WMMA below

        const unsigned short* pAh = sAhi + cur * PLANE;
        const unsigned short* pAl = sAlo + cur * PLANE;
        const unsigned short* pBh = sBhi + cur * PLANE;
        const unsigned short* pBl = sBlo + cur * PLANE;

        Frag16 ahi[4], alo[4], bhi[2], blo[2];
#pragma unroll
        for (int i = 0; i < 4; ++i) {
            const int row = (aRowBase + i * 16) * LDT;
            // A lane: K = {8h..8h+7} then {16+8h..23+8h}
            ahi[i].h[0] = *(const v8bf*)&pAh[row + 8 * half];
            ahi[i].h[1] = *(const v8bf*)&pAh[row + 16 + 8 * half];
            alo[i].h[0] = *(const v8bf*)&pAl[row + 8 * half];
            alo[i].h[1] = *(const v8bf*)&pAl[row + 16 + 8 * half];
        }
#pragma unroll
        for (int j = 0; j < 2; ++j) {
            const int row = (bRowBase + j * 16) * LDT;
            // B lane: K = {8h..8h+7} then {16+8h..23+8h}
            bhi[j].h[0] = *(const v8bf*)&pBh[row + 8 * half];
            bhi[j].h[1] = *(const v8bf*)&pBh[row + 16 + 8 * half];
            blo[j].h[0] = *(const v8bf*)&pBl[row + 8 * half];
            blo[j].h[1] = *(const v8bf*)&pBl[row + 16 + 8 * half];
        }

        // bf16x3: lo*hi + hi*lo + hi*hi
#pragma unroll
        for (int i = 0; i < 4; ++i)
#pragma unroll
            for (int j = 0; j < 2; ++j) {
                acc[i][j] = wmma_bf16(alo[i].v, bhi[j].v, acc[i][j]);
                acc[i][j] = wmma_bf16(ahi[i].v, blo[j].v, acc[i][j]);
                acc[i][j] = wmma_bf16(ahi[i].v, bhi[j].v, acc[i][j]);
            }

        // convert + store NEXT tile into the other buffer (co-issues with WMMA)
        if (more) stash((kt + 1) & 1);
        __syncthreads();
    }

    // ---- epilogue: + bias + SCALE * (t[m,:] . lora_a[:,n]) ----
    // C layout: VGPR v, lanes 0-15 -> (M=v, N=lane); lanes 16-31 -> (M=v+8, N=lane-16)
    const int nA = n0 + waveN * 32 + ln;              // N for j=0; j=1 is nA+16
    const float bn0 = bias[nA];
    const float bn1 = bias[nA + 16];
    float la0[RANK], la1[RANK];
#pragma unroll
    for (int r = 0; r < RANK; ++r) {
        la0[r] = lora_a[(size_t)r * NDIM + nA];
        la1[r] = lora_a[(size_t)r * NDIM + nA + 16];
    }
#pragma unroll
    for (int i = 0; i < 4; ++i) {
        const int mBase = m0 + waveM * 64 + i * 16 + half * 8;
#pragma unroll
        for (int v = 0; v < 8; ++v) {
            const int m = mBase + v;
            const float4 t0 = *(const float4*)(T + (size_t)m * RANK);
            const float4 t1 = *(const float4*)(T + (size_t)m * RANK + 4);
            const float lr0 = t0.x * la0[0] + t0.y * la0[1] + t0.z * la0[2] + t0.w * la0[3]
                            + t1.x * la0[4] + t1.y * la0[5] + t1.z * la0[6] + t1.w * la0[7];
            const float lr1 = t0.x * la1[0] + t0.y * la1[1] + t0.z * la1[2] + t0.w * la1[3]
                            + t1.x * la1[4] + t1.y * la1[5] + t1.z * la1[6] + t1.w * la1[7];
            out[(size_t)m * NDIM + nA]      = acc[i][0][v] + bn0 + LSCALE * lr0;
            out[(size_t)m * NDIM + nA + 16] = acc[i][1][v] + bn1 + LSCALE * lr1;
        }
    }
}

// ---------------------------------------------------------------------------
extern "C" void kernel_launch(void* const* d_in, const int* in_sizes, int n_in,
                              void* d_out, int out_size, void* d_ws, size_t ws_size,
                              hipStream_t stream) {
    const float* x      = (const float*)d_in[0];   // (8*2048, 4096)
    const float* W      = (const float*)d_in[1];   // (4096, 4096)
    const float* b      = (const float*)d_in[2];   // (4096)
    const float* lora_a = (const float*)d_in[3];   // (8, 4096)
    const float* lora_b = (const float*)d_in[4];   // (4096, 8)
    float*       out    = (float*)d_out;
    float*       T      = (float*)d_ws;            // (M, 8) f32 = 512 KB

    const int M = in_sizes[0] / KDIM;              // 16384

    lora_t_kernel<<<M / 8, 256, 0, stream>>>(x, lora_b, T);

    dim3 grid(NDIM / BN, M / BM);                  // (32, 128); N fastest for X reuse in L2
    lora_gemm_kernel<<<grid, 256, 0, stream>>>(x, W, b, lora_a, T, out);
}
